// State_and_Language_Pair_Encoder_23313082483309
// MI455X (gfx1250) — matrix operand order repaired and checked by reading, not compiled, not run on hardware
//
#include <hip/hip_runtime.h>

// ---------------- problem constants ----------------
#define BB    16384
#define SS    256
#define EE    128
#define HH    128
#define NN    34
#define TEMPV 0.1f
#define EPSV  1e-8f

#define TILE_M  32            // batch rows per workgroup
#define MT      (TILE_M / 16) // 16-row M-subtiles per wave
#define THREADS 256           // 8 waves (wave32)

#define CHUNK_ELE 16384       // one 128x128 bf16 weight block (32 KB)
#define NG        (NN * 4)    // total weight chunks: [W1a,W1b,W2,W3] per task

typedef __bf16 bf16_t;
typedef bf16_t v16bf __attribute__((ext_vector_type(16)));
typedef float  v8f   __attribute__((ext_vector_type(8)));

#ifndef __has_builtin
#define __has_builtin(x) 0
#endif
#if __has_builtin(__builtin_amdgcn_tensor_load_to_lds) && \
    __has_builtin(__builtin_amdgcn_s_wait_tensorcnt)
#define HAS_TDM 1
#if __has_include(<hip/amd_detail/amd_gfx1250_TDM.h>)
#define TDM_6ARG 1
#endif
#else
#define HAS_TDM 0
#endif

#if HAS_TDM
typedef unsigned int u32x4 __attribute__((ext_vector_type(4)));
typedef int          i32x4 __attribute__((ext_vector_type(4)));
typedef int          i32x8 __attribute__((ext_vector_type(8)));

// Issue one TDM copy: CHUNK_ELE bf16 (contiguous) global -> LDS.
// D# per ISA 8.3/8.4: group0 = {count=1 | lds_addr | global_addr | type=2},
// group1 = {data_size=8B, tensor_dim0 = tile_dim0 = stride = 4096 x 8B}.
__device__ __forceinline__ void tdm_issue(void* lds_dst, const void* gsrc) {
  const unsigned n8 = (CHUNK_ELE * 2) / 8;  // 4096 eight-byte elements
  const unsigned lds_off = (unsigned)(size_t)lds_dst;   // low 32 bits = LDS offset
  const unsigned long long ga = (unsigned long long)(size_t)gsrc;
  u32x4 g0 = { 1u,                                       // count=1, user mode
               lds_off,                                  // lds_addr
               (unsigned)ga,                             // global_addr[31:0]
               ((unsigned)(ga >> 32) & 0x01FFFFFFu) | 0x80000000u }; // addr[56:32]|type=2
  i32x8 g1 = { (int)(3u << 16),                          // data_size = 8 bytes
               (int)((n8 & 0xFFFFu) << 16),              // tensor_dim0[15:0]
               (int)(n8 >> 16),                          // tensor_dim0[31:16], dim1=0
               (int)((n8 & 0xFFFFu) << 16),              // tile_dim0
               0,                                        // tile_dim1=0, tile_dim2=0
               (int)n8,                                  // tensor_dim0_stride[31:0]
               0, 0 };
  i32x4 gz = {0, 0, 0, 0};
#ifdef TDM_6ARG
  i32x8 gz8 = {0, 0, 0, 0, 0, 0, 0, 0};
  __builtin_amdgcn_tensor_load_to_lds(g0, g1, gz, gz, gz8, 0);
#else
  __builtin_amdgcn_tensor_load_to_lds(g0, g1, gz, gz, 0);
#endif
}
#endif  // HAS_TDM

// ---- A fragment: 16x32 bf16 tile from LDS (row-major, rowStride elements) ----
__device__ __forceinline__ v16bf load_a_frag(const bf16_t* __restrict__ Abase,
                                             int rowStride, int k0, int lane) {
  const int M = lane & 15, half = lane >> 4;
  v16bf a;
#pragma unroll
  for (int i = 0; i < 16; ++i) {
    const int v = i >> 1, p = i & 1;
    const int k = i + half * 16;
    a[i] = Abase[M * rowStride + k0 + k];
  }
  return a;
}

// ---- B fragment from LDS weight tile stored column-major [c][k] (K=128) ----
// lane nc = lane&15 -> column c0+nc; half selects K +16. 16 contiguous bf16.
__device__ __forceinline__ v16bf load_b_frag_lds(const bf16_t* __restrict__ wt,
                                                 int k0, int c0, int lane) {
  const int nc = lane & 15, half = lane >> 4;
  const bf16_t* p = wt + (size_t)(c0 + nc) * HH + k0 + half * 16;
  v16bf b;
#pragma unroll
  for (int i = 0; i < 16; ++i) b[i] = p[i];
  return b;
}

// ---------------- pre-pass: transpose 128x128 fp32 blocks -> bf16 ----------------
// src block z is a 128x128 row-major [k][c] fp32 matrix; destination chunk index
// = ((z >> shift) << 2) + (z & ((1<<shift)-1)) + addc, stored [c][k] bf16.
__global__ __launch_bounds__(256)
void transpose_to_bf16(const float* __restrict__ src, bf16_t* __restrict__ dst,
                       int shift, int addc) {
  __shared__ float tile[32][33];
  const int z = blockIdx.z;
  const int chunk = ((z >> shift) << 2) + (z & ((1 << shift) - 1)) + addc;
  const float* s = src + (size_t)z * HH * HH;
  bf16_t* d = dst + (size_t)chunk * CHUNK_ELE;
  const int k0 = blockIdx.y * 32, c0 = blockIdx.x * 32;
  const int tx = threadIdx.x, ty = threadIdx.y;  // 32 x 8
#pragma unroll
  for (int j = 0; j < 32; j += 8)
    tile[ty + j][tx] = s[(size_t)(k0 + ty + j) * HH + c0 + tx];
  __syncthreads();
#pragma unroll
  for (int j = 0; j < 32; j += 8)
    d[(size_t)(c0 + ty + j) * HH + k0 + tx] = (bf16_t)tile[tx][ty + j];
}

// ---------------- main fused kernel ----------------
__global__ __launch_bounds__(THREADS)
void fused_pair_encoder(const float* __restrict__ state,
                        const int*   __restrict__ task_id,
                        const float* __restrict__ prior,
                        const float* __restrict__ b1,
                        const float* __restrict__ b2,
                        const float* __restrict__ b3,
                        const float* __restrict__ lang_emb,
                        const bf16_t* __restrict__ wsW,   // NG chunks of bf16 weights
                        float* __restrict__ out_rep,      // [B, S+E]
                        float* __restrict__ out_ltp,      // [B, N]
                        float* __restrict__ out_lt,       // [B, E]
                        float* __restrict__ out_lv)       // [B, E]
{
  __shared__ bf16_t wb0[CHUNK_ELE];     // weight ping buffer    32 KB
  __shared__ bf16_t wb1[CHUNK_ELE];     // weight pong buffer    32 KB
  __shared__ bf16_t sA[TILE_M][SS];     // state tile bf16       16 KB
  __shared__ bf16_t sH[TILE_M][HH];     // h1/h2 buffer           8 KB
  __shared__ float  sQ[TILE_M][HH];     // q tile f32            16 KB
  __shared__ float  sLang[TILE_M][EE];  // normalized lang rows  16 KB
  __shared__ float  sLat[TILE_M][EE];   // latent accumulator    16 KB
  __shared__ float  sPpr[TILE_M][NN];
  __shared__ float  sCos[TILE_M][NN];
  __shared__ float  sInvN[TILE_M];
  __shared__ int    sTid[TILE_M];

  const int m0   = blockIdx.x * TILE_M;
  const int t    = threadIdx.x;
  const int lane = t & 31;
  const int wave = t >> 5;
  const int c0   = wave * 16;

#if HAS_TDM
  if (wave == 0) tdm_issue(wb0, wsW);   // prefetch chunk 0 behind the preamble
#endif

  // ---------------- preamble ----------------
  for (int i = t; i < TILE_M * SS; i += THREADS) {
    const int r = i >> 8, c = i & (SS - 1);
    const float v = state[(size_t)(m0 + r) * SS + c];
    sA[r][c] = (bf16_t)v;
    out_rep[(size_t)(m0 + r) * (SS + EE) + c] = v;
  }
  for (int i = t; i < TILE_M * EE; i += THREADS) {
    const int r = i >> 7, c = i & (EE - 1);
    sLat[r][c] = 0.f;
  }
  if (t < TILE_M) {
    const int r  = t;
    const int id = task_id[m0 + r];
    sTid[r] = id;
    float mx = -3.402823466e38f;
    for (int n = 0; n < NN; ++n) mx = fmaxf(mx, prior[(size_t)(m0 + r) * NN + n]);
    float se = 0.f;
    for (int n = 0; n < NN; ++n) {
      const float e = __expf(prior[(size_t)(m0 + r) * NN + n] - mx);
      sPpr[r][n] = e; se += e;
    }
    const float inv = 1.f / se;
    for (int n = 0; n < NN; ++n) sPpr[r][n] *= inv;
    float s = 0.f;
    for (int c = 0; c < EE; ++c) { const float lv = lang_emb[(size_t)id * EE + c]; s += lv * lv; }
    const float invn = 1.f / fmaxf(sqrtf(s), EPSV);
    for (int c = 0; c < EE; ++c) sLang[r][c] = lang_emb[(size_t)id * EE + c] * invn;
  }
  // NOTE: no barrier here; the first acquire() below barriers before use.

  const v8f vzero = {0.f, 0.f, 0.f, 0.f, 0.f, 0.f, 0.f, 0.f};

  // acquire weight chunk g: TDM path waits + issues g+1; fallback copies g.
  auto acquire = [&](int g) -> const bf16_t* {
    bf16_t* cur = (g & 1) ? wb1 : wb0;
#if HAS_TDM
    if (wave == 0) __builtin_amdgcn_s_wait_tensorcnt((short)0);
    __syncthreads();                     // publish chunk g (+ prior stage stores)
    if (wave == 0 && g + 1 < NG) {
      bf16_t* nxt = ((g + 1) & 1) ? wb1 : wb0;
      tdm_issue(nxt, wsW + (size_t)(g + 1) * CHUNK_ELE);
    }
#else
    {
      const uint4* s = (const uint4*)(wsW + (size_t)g * CHUNK_ELE);
      uint4* d = (uint4*)cur;
      for (int i = t; i < (CHUNK_ELE * 2) / 16; i += THREADS) d[i] = s[i];
    }
    __syncthreads();
#endif
    return cur;
  };

  for (int n = 0; n < NN; ++n) {
    const int g = n * 4;

    // ---- stage 1 (two chunks): h1 = relu(state @ W1 + b1), K = 256 ----
    v8f acc[MT];
#pragma unroll
    for (int m = 0; m < MT; ++m) acc[m] = vzero;

    {
      const bf16_t* wc = acquire(g);            // W1a (k = 0..127)
#pragma unroll
      for (int kk = 0; kk < 4; ++kk) {
        const v16bf bfr = load_b_frag_lds(wc, kk * 32, c0, lane);
#pragma unroll
        for (int m = 0; m < MT; ++m) {
          const v16bf afr = load_a_frag(&sA[16 * m][0], SS, kk * 32, lane);
          acc[m] = __builtin_amdgcn_wmma_f32_16x16x32_bf16(
              false, afr, false, bfr, (short)0, acc[m], false, false);
        }
      }
    }
    {
      const bf16_t* wc = acquire(g + 1);        // W1b (k = 128..255)
#pragma unroll
      for (int kk = 0; kk < 4; ++kk) {
        const v16bf bfr = load_b_frag_lds(wc, kk * 32, c0, lane);
#pragma unroll
        for (int m = 0; m < MT; ++m) {
          const v16bf afr = load_a_frag(&sA[16 * m][0], SS, 128 + kk * 32, lane);
          acc[m] = __builtin_amdgcn_wmma_f32_16x16x32_bf16(
              false, afr, false, bfr, (short)0, acc[m], false, false);
        }
      }
      const int half = lane >> 4, nc = lane & 15;
#pragma unroll
      for (int m = 0; m < MT; ++m)
#pragma unroll
        for (int i = 0; i < 8; ++i) {
          const int row = 16 * m + i + 8 * half, col = c0 + nc;
          sH[row][col] = (bf16_t)fmaxf(acc[m][i] + b1[(size_t)n * HH + col], 0.f);
        }
    }

    // ---- stage 2: h2 = relu(h1 @ W2 + b2), K = 128, in-place sH ----
    {
      const bf16_t* wc = acquire(g + 2);        // barrier publishes sH + W2
      v8f a2[MT];
#pragma unroll
      for (int m = 0; m < MT; ++m) a2[m] = vzero;
#pragma unroll
      for (int kk = 0; kk < 4; ++kk) {
        const v16bf bfr = load_b_frag_lds(wc, kk * 32, c0, lane);
#pragma unroll
        for (int m = 0; m < MT; ++m) {
          const v16bf afr = load_a_frag(&sH[16 * m][0], HH, kk * 32, lane);
          a2[m] = __builtin_amdgcn_wmma_f32_16x16x32_bf16(
              false, afr, false, bfr, (short)0, a2[m], false, false);
        }
      }
      __syncthreads();                          // all waves done READING sH
      const int half = lane >> 4, nc = lane & 15;
#pragma unroll
      for (int m = 0; m < MT; ++m)
#pragma unroll
        for (int i = 0; i < 8; ++i) {
          const int row = 16 * m + i + 8 * half, col = c0 + nc;
          sH[row][col] = (bf16_t)fmaxf(a2[m][i] + b2[(size_t)n * HH + col], 0.f);
        }
    }

    // ---- stage 3: q = h2 @ W3 + b3, K = 128 ----
    {
      const bf16_t* wc = acquire(g + 3);        // barrier publishes sH + W3
      v8f a3[MT];
#pragma unroll
      for (int m = 0; m < MT; ++m) a3[m] = vzero;
#pragma unroll
      for (int kk = 0; kk < 4; ++kk) {
        const v16bf bfr = load_b_frag_lds(wc, kk * 32, c0, lane);
#pragma unroll
        for (int m = 0; m < MT; ++m) {
          const v16bf afr = load_a_frag(&sH[16 * m][0], HH, kk * 32, lane);
          a3[m] = __builtin_amdgcn_wmma_f32_16x16x32_bf16(
              false, afr, false, bfr, (short)0, a3[m], false, false);
        }
      }
      const int half = lane >> 4, nc = lane & 15;
#pragma unroll
      for (int m = 0; m < MT; ++m)
#pragma unroll
        for (int i = 0; i < 8; ++i) {
          const int row = 16 * m + i + 8 * half, col = c0 + nc;
          sQ[row][col] = a3[m][i] + b3[(size_t)n * EE + col];
        }
    }
    __syncthreads();

    // ---- epilogue: normalize q, cosine, latent accumulation ----
    if (t < TILE_M) {
      const int r = t;
      float s = 0.f;
      for (int c = 0; c < EE; ++c) { const float qv = sQ[r][c]; s += qv * qv; }
      const float inv = rsqrtf(s);
      sInvN[r] = inv;
      float cs = 0.f;
      for (int c = 0; c < EE; ++c) cs += sQ[r][c] * sLang[r][c];
      sCos[r][n] = cs * inv;
    }
    __syncthreads();
    for (int i = t; i < TILE_M * EE; i += THREADS) {
      const int r = i >> 7, c = i & (EE - 1);
      const float qn = sQ[r][c] * sInvN[r];
      sLat[r][c] += qn * sPpr[r][n];
      if (sTid[r] == n) out_lt[(size_t)(m0 + r) * EE + c] = qn;
    }
    __syncthreads();
  }

  // ---------------- finalize ----------------
  if (t < TILE_M) {
    const int r = t;
    const float invT = 1.f / TEMPV;
    float mx = -3.402823466e38f;
    for (int n = 0; n < NN; ++n) mx = fmaxf(mx, sCos[r][n]);
    float se = 0.f;
    for (int n = 0; n < NN; ++n) se += __expf((sCos[r][n] - mx) * invT);
    const float lse = mx * invT + __logf(se);
    for (int n = 0; n < NN; ++n)
      out_ltp[(size_t)(m0 + r) * NN + n] = sCos[r][n] * invT - lse;
  }
  for (int i = t; i < TILE_M * EE; i += THREADS) {
    const int r = i >> 7, c = i & (EE - 1);
    const float v = sLat[r][c];
    out_lv[(size_t)(m0 + r) * EE + c] = v;
    out_rep[(size_t)(m0 + r) * (SS + EE) + SS + c] = v;
  }
}

extern "C" void kernel_launch(void* const* d_in, const int* in_sizes, int n_in,
                              void* d_out, int out_size, void* d_ws, size_t ws_size,
                              hipStream_t stream) {
  const float* state   = (const float*)d_in[0];
  const int*   task_id = (const int*)  d_in[1];
  const float* prior   = (const float*)d_in[2];
  const float* W1      = (const float*)d_in[3];
  const float* b1v     = (const float*)d_in[4];
  const float* W2      = (const float*)d_in[5];
  const float* b2v     = (const float*)d_in[6];
  const float* W3      = (const float*)d_in[7];
  const float* b3v     = (const float*)d_in[8];
  const float* lang    = (const float*)d_in[9];

  bf16_t* wsW = (bf16_t*)d_ws;  // NG * CHUNK_ELE bf16 = 4.46 MB

  // pre-pass: transpose+convert each 128x128 fp32 block into its chunk slot
  {
    dim3 blk(32, 8);
    dim3 g1(4, 4, NN * 2);  // W1: blocks z -> chunk (z>>1)*4 + (z&1)
    transpose_to_bf16<<<g1, blk, 0, stream>>>(W1, wsW, 1, 0);
    dim3 g2(4, 4, NN);      // W2: chunk z*4 + 2
    transpose_to_bf16<<<g2, blk, 0, stream>>>(W2, wsW, 0, 2);
    dim3 g3(4, 4, NN);      // W3: chunk z*4 + 3
    transpose_to_bf16<<<g3, blk, 0, stream>>>(W3, wsW, 0, 3);
  }

  float* out     = (float*)d_out;
  float* out_rep = out;                               // [B, S+E]
  float* out_ltp = out_rep + (size_t)BB * (SS + EE);  // [B, N]
  float* out_lt  = out_ltp + (size_t)BB * NN;         // [B, E]
  float* out_lv  = out_lt  + (size_t)BB * EE;         // [B, E]

  dim3 grid(BB / TILE_M);
  fused_pair_encoder<<<grid, THREADS, 0, stream>>>(
      state, task_id, prior, b1v, b2v, b3v, lang, wsW,
      out_rep, out_ltp, out_lt, out_lv);
}
